// VectorQuantizer_81621558493560
// MI455X (gfx1250) — compile-verified, matrix operand order checked by reading, not run on hardware
//
#include <hip/hip_runtime.h>
#include <hip/hip_bf16.h>

typedef __attribute__((ext_vector_type(16))) _Float16 v16h;
typedef __attribute__((ext_vector_type(8)))  _Float16 v8h;
typedef __attribute__((ext_vector_type(8)))  float    v8f;

#define VQ_N   131072L        // B*H*W = 32*64*64
#define VQ_C   64
#define VQ_K   1024
#define VQ_Q   8388608L       // B*C*H*W
#define VQ_HW  4096           // H*W

// ---------------------------------------------------------------------------
// Kernel 1: codebook prep — f32 -> f16 copy + per-code squared norms (f32)
// ---------------------------------------------------------------------------
__global__ __launch_bounds__(256) void vq_prep(const float* __restrict__ emb,
                                               _Float16* __restrict__ embh,
                                               float* __restrict__ embnorm) {
    int k = blockIdx.x * blockDim.x + threadIdx.x;   // 0..1023, one code per thread
    if (k >= VQ_K) return;
    const float* row = emb + (long)k * VQ_C;
    _Float16* dst = embh + (long)k * VQ_C;
    float s = 0.0f;
#pragma unroll
    for (int c = 0; c < VQ_C; ++c) {
        float v = row[c];
        s += v * v;
        dst[c] = (_Float16)v;
    }
    embnorm[k] = s;
}

// ---------------------------------------------------------------------------
// Kernel 2: fused (flat @ emb^T) + argmin + gather.
//   One wave32 handles 32 rows (two 16-row M-tiles sharing B fragments).
//   flat[n][c] = x[b][c][hw]  (n = b*4096 + hw), stride 4096 along c.
//   Per code-tile: 4 x v_wmma_f32_16x16x32_f16 in two independent chains.
// ---------------------------------------------------------------------------
__global__ __launch_bounds__(32) void vq_main(const float* __restrict__ x,
                                              const float* __restrict__ emb,
                                              const _Float16* __restrict__ embh,
                                              const float* __restrict__ embnorm,
                                              float* __restrict__ out_q,    // [N*64]
                                              float* __restrict__ out_idx)  // [N]
{
    __shared__ float xtile[32][VQ_C + 1];   // [n][c], pad to dodge bank conflicts
    __shared__ float bests[32][16];
    __shared__ int   besti[32][16];
    __shared__ int   idxrow[32];

    const int  lane  = threadIdx.x;               // 0..31
    const long nbase = (long)blockIdx.x * 32;     // 32 | 4096 -> never crosses b
    const long bimg  = nbase >> 12;               // image index
    const int  hw0   = (int)(nbase & (VQ_HW - 1));
    const float* xb  = x + bimg * (long)VQ_C * VQ_HW + hw0;

    // ---- coalesced x-tile load: each iter = one 128B-contiguous segment ----
#pragma unroll
    for (int c = 0; c < VQ_C; ++c) {
        xtile[lane][c] = xb[(long)c * VQ_HW + lane];
    }
    __syncthreads();

    // ---- A fragments (ISA 16-bit A layout: lane=M, two 8-run K halves) ----
    const int m  = lane & 15;
    const int cb = (lane >> 4) * 8;       // k-sub-base per lane half
    union { _Float16 h[16]; v16h v; } A0a, A1a, A0b, A1b;
#pragma unroll
    for (int j = 0; j < 8; ++j) {
        A0a.h[j]     = (_Float16)xtile[m][cb + j];
        A0a.h[j + 8] = (_Float16)xtile[m][cb + 16 + j];
        A1a.h[j]     = (_Float16)xtile[m][32 + cb + j];
        A1a.h[j + 8] = (_Float16)xtile[m][32 + cb + 16 + j];
        A0b.h[j]     = (_Float16)xtile[m + 16][cb + j];
        A0b.h[j + 8] = (_Float16)xtile[m + 16][cb + 16 + j];
        A1b.h[j]     = (_Float16)xtile[m + 16][32 + cb + j];
        A1b.h[j + 8] = (_Float16)xtile[m + 16][32 + cb + 16 + j];
    }

    float best_sa[8], best_sb[8];
    int   best_ia[8], best_ib[8];
#pragma unroll
    for (int v = 0; v < 8; ++v) {
        best_sa[v] = 3.4e38f; best_ia[v] = 0;
        best_sb[v] = 3.4e38f; best_ib[v] = 0;
    }

    // ---- sweep the 1024 codes in tiles of 16 ----
    for (int t = 0; t < VQ_K / 16; ++t) {
        const int mycode = t * 16 + m;                       // B-frag col = lane&15
        const _Float16* erow = embh + (long)mycode * VQ_C;   // 128B-aligned row

        v8h lo0 = *(const v8h*)(erow + cb);
        v8h hi0 = *(const v8h*)(erow + cb + 16);
        v8h lo1 = *(const v8h*)(erow + 32 + cb);
        v8h hi1 = *(const v8h*)(erow + 32 + cb + 16);
        v16h B0 = __builtin_shufflevector(lo0, hi0, 0,1,2,3,4,5,6,7,8,9,10,11,12,13,14,15);
        v16h B1 = __builtin_shufflevector(lo1, hi1, 0,1,2,3,4,5,6,7,8,9,10,11,12,13,14,15);

        // two independent accumulator chains -> scheduler hides WMMA->VALU NOPs
        v8f acca = {};
        v8f accb = {};
        acca = __builtin_amdgcn_wmma_f32_16x16x32_f16(false, A0a.v, false, B0,
                                                      (short)0, acca, false, false);
        accb = __builtin_amdgcn_wmma_f32_16x16x32_f16(false, A0b.v, false, B0,
                                                      (short)0, accb, false, false);
        acca = __builtin_amdgcn_wmma_f32_16x16x32_f16(false, A1a.v, false, B1,
                                                      (short)0, acca, false, false);
        accb = __builtin_amdgcn_wmma_f32_16x16x32_f16(false, A1b.v, false, B1,
                                                      (short)0, accb, false, false);

        const float en = embnorm[mycode];
#pragma unroll
        for (int v = 0; v < 8; ++v) {
            float sa = en - 2.0f * acca[v];    // d2 minus the row-constant |x|^2
            if (sa < best_sa[v]) { best_sa[v] = sa; best_ia[v] = mycode; }
            float sb = en - 2.0f * accb[v];
            if (sb < best_sb[v]) { best_sb[v] = sb; best_ib[v] = mycode; }
        }
    }

    // ---- cross-lane argmin: C/D layout row = v + (lane>=16 ? 8 : 0) ----
    {
        const int rbase = (lane >> 4) * 8;
#pragma unroll
        for (int v = 0; v < 8; ++v) {
            bests[rbase + v][m]      = best_sa[v];
            besti[rbase + v][m]      = best_ia[v];
            bests[16 + rbase + v][m] = best_sb[v];
            besti[16 + rbase + v][m] = best_ib[v];
        }
    }
    __syncthreads();
    {
        float bs = bests[lane][0];
        int   bi = besti[lane][0];
#pragma unroll
        for (int c2 = 1; c2 < 16; ++c2) {
            float s  = bests[lane][c2];
            int   ii = besti[lane][c2];
            if (s < bs || (s == bs && ii < bi)) { bs = s; bi = ii; }  // first-min ties
        }
        idxrow[lane] = bi;
        out_idx[nbase + lane] = (float)bi;
    }
    __syncthreads();

    // ---- gather quantized = emb[idx], flat row-major [N][C] (= buggy reshape)
#pragma unroll 4
    for (int i = 0; i < 32; ++i) {
        const int code = idxrow[i];
        const float2 val = *(const float2*)(emb + (long)code * VQ_C + 2 * lane);
        *(float2*)(out_q + (nbase + i) * VQ_C + 2 * lane) = val;
    }
}

// ---------------------------------------------------------------------------
// Kernels 3/4: deterministic two-pass loss reduction (no float atomics)
// ---------------------------------------------------------------------------
__global__ __launch_bounds__(256) void vq_loss_partial(const float* __restrict__ x,
                                                       const float* __restrict__ q,
                                                       float* __restrict__ partials) {
    __shared__ float red[256];
    const long total4 = VQ_Q / 4;
    long tid    = (long)blockIdx.x * blockDim.x + threadIdx.x;
    long stride = (long)gridDim.x * blockDim.x;
    const float4* x4 = (const float4*)x;
    const float4* q4 = (const float4*)q;
    float s = 0.0f;
    for (long i = tid; i < total4; i += stride) {
        float4 a = x4[i], b = q4[i];
        float d0 = b.x - a.x, d1 = b.y - a.y, d2 = b.z - a.z, d3 = b.w - a.w;
        s += d0 * d0 + d1 * d1 + d2 * d2 + d3 * d3;
    }
    red[threadIdx.x] = s;
    __syncthreads();
    for (int off = 128; off > 0; off >>= 1) {
        if ((int)threadIdx.x < off) red[threadIdx.x] += red[threadIdx.x + off];
        __syncthreads();
    }
    if (threadIdx.x == 0) partials[blockIdx.x] = red[0];
}

__global__ __launch_bounds__(256) void vq_loss_final(const float* __restrict__ partials,
                                                     float* __restrict__ loss) {
    __shared__ float red[256];
    float s = 0.0f;
    for (int i = threadIdx.x; i < 1024; i += 256) s += partials[i];
    red[threadIdx.x] = s;
    __syncthreads();
    for (int off = 128; off > 0; off >>= 1) {
        if ((int)threadIdx.x < off) red[threadIdx.x] += red[threadIdx.x + off];
        __syncthreads();
    }
    // loss = 0.25*mean((sg(q)-x)^2) + mean((q-sg(x))^2) = 1.25 * mean((q-x)^2)
    if (threadIdx.x == 0) *loss = red[0] * (1.25f / (float)VQ_Q);
}

// ---------------------------------------------------------------------------
extern "C" void kernel_launch(void* const* d_in, const int* in_sizes, int n_in,
                              void* d_out, int out_size, void* d_ws, size_t ws_size,
                              hipStream_t stream) {
    const float* x   = (const float*)d_in[0];   // [32,64,64,64] f32
    const float* emb = (const float*)d_in[1];   // [1024,64] f32

    float* out      = (float*)d_out;
    float* out_q    = out;               // [8388608] quantized_ste (== quantized)
    float* out_loss = out + VQ_Q;        // [1]
    float* out_idx  = out + VQ_Q + 1;    // [131072] indices (as float values)

    // workspace layout
    _Float16* embh    = (_Float16*)d_ws;                         // 131072 B
    float*    embnorm = (float*)((char*)d_ws + 131072);          //   4096 B
    float*    partials= (float*)((char*)d_ws + 131072 + 4096);   //   4096 B

    vq_prep<<<4, 256, 0, stream>>>(emb, embh, embnorm);
    vq_main<<<(int)(VQ_N / 32), 32, 0, stream>>>(x, emb, embh, embnorm,
                                                 out_q, out_idx);
    vq_loss_partial<<<1024, 256, 0, stream>>>(x, out_q, partials);
    vq_loss_final<<<1, 256, 0, stream>>>(partials, out_loss);
}
